// DeltaNetBlock_34342558498950
// MI455X (gfx1250) — matrix operand order repaired and checked by reading, not compiled, not run on hardware
//
#include <hip/hip_runtime.h>
#include <hip/hip_bf16.h>
#include <math.h>

typedef __attribute__((ext_vector_type(16))) _Float16 v16h;
typedef __attribute__((ext_vector_type(8)))  _Float16 v8h;
typedef __attribute__((ext_vector_type(8)))  float    v8f;

#define B_ 4
#define L_ 4096
#define H_ 768
#define E_ 1536

// ---------------------------------------------------------------------------
// CDNA5 async global->LDS copy (16B per lane) + wait, tracked by ASYNCcnt.
// Inline asm per CDNA5 ISA (portable across toolchains).
// ---------------------------------------------------------------------------
__device__ __forceinline__ void async_copy16(_Float16* lds, const _Float16* g) {
  unsigned lds_off = (unsigned)(uintptr_t)lds;  // low 32 bits = LDS offset
  asm volatile("global_load_async_to_lds_b128 %0, %1, off"
               :
               : "v"(lds_off), "v"(g)
               : "memory");
}
__device__ __forceinline__ void wait_async0() {
  asm volatile("s_wait_asynccnt 0" ::: "memory");
}

// ---------------------------------------------------------------------------
// block-wide sum reduction (256 threads)
// ---------------------------------------------------------------------------
__device__ __forceinline__ float block_reduce_sum(float v, float* sm) {
  int tid = threadIdx.x;
  sm[tid] = v;
  __syncthreads();
  for (int s = 128; s > 0; s >>= 1) {
    if (tid < s) sm[tid] += sm[tid + s];
    __syncthreads();
  }
  float r = sm[0];
  __syncthreads();
  return r;
}

// ---------------------------------------------------------------------------
// LayerNorm (optionally pre-scaled) -> f16 output.  One block per row.
// ---------------------------------------------------------------------------
__global__ __launch_bounds__(256) void ln_to_f16_kernel(
    const float* __restrict__ x, const float* __restrict__ g,
    const float* __restrict__ b, const float* __restrict__ scale,
    _Float16* __restrict__ y, int C) {
  __shared__ float sm[256];
  size_t row = blockIdx.x;
  const float* xr = x + row * (size_t)C;
  float s = scale ? scale[0] : 1.0f;
  float acc = 0.f;
  for (int c = threadIdx.x; c < C; c += 256) acc += xr[c] * s;
  float mean = block_reduce_sum(acc, sm) / (float)C;
  float accv = 0.f;
  for (int c = threadIdx.x; c < C; c += 256) {
    float d = xr[c] * s - mean;
    accv += d * d;
  }
  float var = block_reduce_sum(accv, sm) / (float)C;
  float inv = rsqrtf(var + 1e-5f);
  for (int c = threadIdx.x; c < C; c += 256) {
    float v = (xr[c] * s - mean) * inv * g[c] + b[c];
    y[row * (size_t)C + c] = (_Float16)v;
  }
}

// ---------------------------------------------------------------------------
// WMMA GEMM with async double-buffered LDS staging:
//   C[M,N] = A[M,K] * W[N,K]^T + bias + residual
// A, W f16 row-major; C f32.  Block = 256 threads = 8 waves.
// Block tile 128x128, wave tile 64x32 (4x2 WMMA 16x16x32), K-step 32.
// LDS: 2 x (128x40 halves) per operand = 40 KB, rows padded for alignment.
// Requires M%128==0, N%128==0, K%32==0 (true for every matmul here).
// ---------------------------------------------------------------------------
#define ROWH 40  // 32 K-halves + 8 pad halves (80 B row stride, 16B aligned)

__global__ __launch_bounds__(256) void gemm_wmma_kernel(
    const _Float16* __restrict__ A, const _Float16* __restrict__ W,
    const float* __restrict__ bias, const float* __restrict__ res,
    float* __restrict__ C, int M, int N, int K) {
  __shared__ _Float16 As[2][128 * ROWH];
  __shared__ _Float16 Bs[2][128 * ROWH];

  const int tid   = threadIdx.x;
  const int lane  = tid & 31;
  const int wave  = tid >> 5;
  const int waveM = wave & 1;   // 2 waves along M
  const int waveN = wave >> 1;  // 4 waves along N
  const int lhalf = lane >> 4;  // 0 or 1
  const int lmod  = lane & 15;
  const int aKoff = lhalf * 8;   // A frag: K halves {0..7,16..23} / {8..15,24..31}
  const int bKoff = lhalf * 8;   // B frag: K halves {0..7,16..23} / {8..15,24..31}
  const int blockM = blockIdx.x * 128;
  const int blockN = blockIdx.y * 128;
  const int wmL = waveM * 64;    // wave tile origin inside block tile
  const int wnL = waveN * 32;

  // cooperative copy map: 512 chunks of 16B per 128x32 tile, 2 per thread
  const int c0 = tid, c1 = tid + 256;
  const int ar0 = c0 >> 2, ac0 = (c0 & 3) * 8;
  const int ar1 = c1 >> 2, ac1 = (c1 & 3) * 8;

  const _Float16* Ag = A + (size_t)blockM * K;
  const _Float16* Bg = W + (size_t)blockN * K;

  v8f acc[4][2] = {};

  // prologue: start async fill of buffer 0
  {
    async_copy16(&As[0][ar0 * ROWH + ac0], Ag + (size_t)ar0 * K + ac0);
    async_copy16(&As[0][ar1 * ROWH + ac1], Ag + (size_t)ar1 * K + ac1);
    async_copy16(&Bs[0][ar0 * ROWH + ac0], Bg + (size_t)ar0 * K + ac0);
    async_copy16(&Bs[0][ar1 * ROWH + ac1], Bg + (size_t)ar1 * K + ac1);
  }

  const int nK = K >> 5;
  for (int kb = 0; kb < nK; kb++) {
    const int cur = kb & 1;
    wait_async0();     // our async fills of buf[cur] have landed in LDS
    __syncthreads();   // everyone's fills visible; prior reads of buf[cur^1] done

    if (kb + 1 < nK) {  // overlap next slab's DMA with this slab's WMMAs
      const int nxt = cur ^ 1;
      const int k = (kb + 1) << 5;
      async_copy16(&As[nxt][ar0 * ROWH + ac0], Ag + (size_t)ar0 * K + k + ac0);
      async_copy16(&As[nxt][ar1 * ROWH + ac1], Ag + (size_t)ar1 * K + k + ac1);
      async_copy16(&Bs[nxt][ar0 * ROWH + ac0], Bg + (size_t)ar0 * K + k + ac0);
      async_copy16(&Bs[nxt][ar1 * ROWH + ac1], Bg + (size_t)ar1 * K + k + ac1);
    }

    v16h af[4], bf[2];
#pragma unroll
    for (int tm = 0; tm < 4; tm++) {
      const _Float16* p = &As[cur][(wmL + tm * 16 + lmod) * ROWH + aKoff];
      v8h lo = *(const v8h*)p;
      v8h hi = *(const v8h*)(p + 16);
      af[tm] = __builtin_shufflevector(lo, hi, 0, 1, 2, 3, 4, 5, 6, 7,
                                       8, 9, 10, 11, 12, 13, 14, 15);
    }
#pragma unroll
    for (int tn = 0; tn < 2; tn++) {
      const _Float16* p = &Bs[cur][(wnL + tn * 16 + lmod) * ROWH + bKoff];
      v8h lo = *(const v8h*)p;
      v8h hi = *(const v8h*)(p + 16);
      bf[tn] = __builtin_shufflevector(lo, hi, 0, 1, 2, 3, 4, 5, 6, 7,
                                       8, 9, 10, 11, 12, 13, 14, 15);
    }
#pragma unroll
    for (int tm = 0; tm < 4; tm++)
#pragma unroll
      for (int tn = 0; tn < 2; tn++)
        acc[tm][tn] = __builtin_amdgcn_wmma_f32_16x16x32_f16(
            false, af[tm], false, bf[tn], (short)0, acc[tm][tn], false, false);
  }

  const int rbase = lhalf * 8;  // C/D layout: vgpr r -> M = r + 8*(lane>=16)
#pragma unroll
  for (int tm = 0; tm < 4; tm++) {
#pragma unroll
    for (int tn = 0; tn < 2; tn++) {
      int n = blockN + wnL + tn * 16 + lmod;
      float bv = bias ? bias[n] : 0.f;
#pragma unroll
      for (int r = 0; r < 8; r++) {
        int m = blockM + wmL + tm * 16 + rbase + r;
        size_t idx = (size_t)m * N + n;
        float v = acc[tm][tn][r] + bv;
        if (res) v += res[idx];
        C[idx] = v;
      }
    }
  }
}

// ---------------------------------------------------------------------------
// Activation kernel (in-place on qkv rows): silu(q), silu(k), gelu(v),
// q = l2n(q) + 0.1*k ; k = l2n(k) + 0.1*q_new.  One block per token.
// ---------------------------------------------------------------------------
__global__ __launch_bounds__(256) void act_kernel(float* __restrict__ qkv) {
  __shared__ float sm[256];
  constexpr int NIT = E_ / 256;  // 6
  size_t t = blockIdx.x;
  float* q = qkv + t * (size_t)(3 * E_);
  float* k = q + E_;
  float* v = k + E_;
  float sq[NIT], sk[NIT];
  float ssq = 0.f, ssk = 0.f;
#pragma unroll
  for (int i = 0; i < NIT; i++) {
    int e = threadIdx.x + i * 256;
    float qs = q[e], ks = k[e], vv = v[e];
    float a = qs / (1.f + expf(-qs));   // silu
    float c = ks / (1.f + expf(-ks));   // silu
    sq[i] = a; sk[i] = c;
    ssq += a * a; ssk += c * c;
    v[e] = 0.5f * vv * (1.f + erff(vv * 0.70710678118654752f));  // exact gelu
  }
  float nq = sqrtf(block_reduce_sum(ssq, sm));
  float nk = sqrtf(block_reduce_sum(ssk, sm));
  float iq = 1.f / fmaxf(nq, 1e-12f);
  float ik = 1.f / fmaxf(nk, 1e-12f);
#pragma unroll
  for (int i = 0; i < NIT; i++) {
    int e = threadIdx.x + i * 256;
    float qn = sq[i] * iq + 0.1f * sk[i];
    float kn = sk[i] * ik + 0.1f * qn;   // uses updated q, like the reference
    q[e] = qn;
    k[e] = kn;
  }
}

// ---------------------------------------------------------------------------
// Depthwise conv (kernel 3, pad 1, along L) + beta gating + f16 packing.
//   q      -> qF16 [b][l][e]   (row-major, GEMM A operand)
//   k      -> kT   [b][e][l]   (transposed, GEMM "weight" operand)
//   beta*v -> vT   [b][e][l]   (transposed)
// ---------------------------------------------------------------------------
__device__ __forceinline__ float conv3(const float* __restrict__ qkv, size_t t,
                                       int l, size_t off, float w0, float w1,
                                       float w2) {
  const size_t stride = (size_t)(3 * E_);
  float xm = (l > 0)      ? qkv[(t - 1) * stride + off] : 0.f;
  float x0 =                qkv[t * stride + off];
  float xp = (l < L_ - 1) ? qkv[(t + 1) * stride + off] : 0.f;
  return w0 * xm + w1 * x0 + w2 * xp;
}

__global__ __launch_bounds__(256) void conv_pack_kernel(
    const float* __restrict__ qkv, const float* __restrict__ betaLin,
    const float* __restrict__ conv_w, _Float16* __restrict__ qF16,
    _Float16* __restrict__ kT, _Float16* __restrict__ vT) {
  int l = blockIdx.x;
  int b = blockIdx.y;
  size_t t = (size_t)b * L_ + l;
  for (int e = threadIdx.x; e < E_; e += 256) {
    float w0 = conv_w[e * 3 + 0];
    float w1 = conv_w[e * 3 + 1];
    float w2 = conv_w[e * 3 + 2];
    float yq = conv3(qkv, t, l, (size_t)0 * E_ + e, w0, w1, w2);
    float yk = conv3(qkv, t, l, (size_t)1 * E_ + e, w0, w1, w2);
    float yv = conv3(qkv, t, l, (size_t)2 * E_ + e, w0, w1, w2);
    float bl = betaLin[t * (size_t)E_ + e];
    float beta = 1.f / (1.f + expf(-bl)) * 0.9f + 0.1f;
    qF16[t * (size_t)E_ + e] = (_Float16)yq;
    size_t tidx = ((size_t)b * E_ + e) * L_ + l;
    kT[tidx] = (_Float16)yk;
    vT[tidx] = (_Float16)(beta * yv);
  }
}

// ---------------------------------------------------------------------------
// elementwise helpers
// ---------------------------------------------------------------------------
__global__ __launch_bounds__(256) void cast_f16_kernel(
    const float* __restrict__ s, _Float16* __restrict__ d, size_t n) {
  for (size_t i = (size_t)blockIdx.x * 256 + threadIdx.x; i < n;
       i += (size_t)gridDim.x * 256)
    d[i] = (_Float16)s[i];
}

__global__ __launch_bounds__(256) void gelu_f16_kernel(
    const float* __restrict__ s, _Float16* __restrict__ d, size_t n) {
  for (size_t i = (size_t)blockIdx.x * 256 + threadIdx.x; i < n;
       i += (size_t)gridDim.x * 256) {
    float x = s[i];
    d[i] = (_Float16)(0.5f * x * (1.f + erff(x * 0.70710678118654752f)));
  }
}

// ---------------------------------------------------------------------------
// host-side orchestration
// ---------------------------------------------------------------------------
extern "C" void kernel_launch(void* const* d_in, const int* in_sizes, int n_in,
                              void* d_out, int out_size, void* d_ws,
                              size_t ws_size, hipStream_t stream) {
  (void)in_sizes; (void)n_in; (void)out_size; (void)ws_size;
  const float* x       = (const float*)d_in[0];
  const float* ln1_g   = (const float*)d_in[1];
  const float* ln1_b   = (const float*)d_in[2];
  const float* ln2_g   = (const float*)d_in[3];
  const float* ln2_b   = (const float*)d_in[4];
  const float* W_in    = (const float*)d_in[5];
  const float* b_in    = (const float*)d_in[6];
  const float* W_beta  = (const float*)d_in[7];
  const float* b_beta  = (const float*)d_in[8];
  const float* W_out   = (const float*)d_in[9];
  const float* b_out   = (const float*)d_in[10];
  const float* W1      = (const float*)d_in[11];
  const float* b1      = (const float*)d_in[12];
  const float* W2      = (const float*)d_in[13];
  const float* b2      = (const float*)d_in[14];
  const float* conv_w  = (const float*)d_in[15];
  const float* a_scale = (const float*)d_in[16];
  float* out = (float*)d_out;

  const size_t T = (size_t)B_ * L_;  // 16384 tokens

  char* ws = (char*)d_ws;
  size_t off = 0;
  auto alloc = [&](size_t bytes) -> char* {
    char* p = ws + off;
    off = (off + bytes + 255) & ~(size_t)255;
    return p;
  };

  _Float16* xnF16 = (_Float16*)alloc(T * H_ * 2);       // later: hnF16 (MLP LN)
  _Float16* WinH  = (_Float16*)alloc((size_t)3 * E_ * H_ * 2);
  _Float16* WbH   = (_Float16*)alloc((size_t)E_ * H_ * 2);
  _Float16* WoH   = (_Float16*)alloc((size_t)H_ * E_ * 2);
  _Float16* W1H   = (_Float16*)alloc((size_t)4 * H_ * H_ * 2);
  _Float16* W2H   = (_Float16*)alloc((size_t)H_ * 4 * H_ * 2);
  float* qkv      = (float*)alloc(T * 3 * E_ * 4);      // later: m1 (MLP hidden)
  float* betaLin  = (float*)alloc(T * E_ * 4);          // later: attn, then geluF16
  _Float16* qF16  = (_Float16*)alloc(T * E_ * 2);       // later: yF16 (post-LN2)
  _Float16* kT    = (_Float16*)alloc(T * E_ * 2);
  _Float16* vT    = (_Float16*)alloc(T * E_ * 2);
  float* stateF32 = (float*)alloc((size_t)B_ * E_ * E_ * 4);
  _Float16* stateF16 = (_Float16*)alloc((size_t)B_ * E_ * E_ * 2);
  float* x2       = (float*)alloc(T * H_ * 4);          // residual after attn

  float* m1 = qkv;                         // [T, 4H] f32, reuse
  float* attn = betaLin;                   // [T, E]  f32, reuse
  _Float16* geluF16 = (_Float16*)betaLin;  // [T, 4H] f16, reuse (same bytes)
  _Float16* yF16 = qF16;                   // [T, E] f16, reuse
  _Float16* hnF16 = xnF16;                 // [T, H] f16, reuse

  // 1. xn = LN(x) (f16)
  ln_to_f16_kernel<<<(int)T, 256, 0, stream>>>(x, ln1_g, ln1_b, nullptr, xnF16, H_);

  // 2. weights -> f16
  cast_f16_kernel<<<2048, 256, 0, stream>>>(W_in,  WinH, (size_t)3 * E_ * H_);
  cast_f16_kernel<<<2048, 256, 0, stream>>>(W_beta, WbH, (size_t)E_ * H_);
  cast_f16_kernel<<<2048, 256, 0, stream>>>(W_out,  WoH, (size_t)H_ * E_);
  cast_f16_kernel<<<2048, 256, 0, stream>>>(W1,     W1H, (size_t)4 * H_ * H_);
  cast_f16_kernel<<<2048, 256, 0, stream>>>(W2,     W2H, (size_t)H_ * 4 * H_);

  // 3. qkv = xn @ W_in^T + b_in  ;  betaLin = xn @ W_beta^T + b_beta
  gemm_wmma_kernel<<<dim3((int)T / 128, (3 * E_) / 128), 256, 0, stream>>>(
      xnF16, WinH, b_in, nullptr, qkv, (int)T, 3 * E_, H_);
  gemm_wmma_kernel<<<dim3((int)T / 128, E_ / 128), 256, 0, stream>>>(
      xnF16, WbH, b_beta, nullptr, betaLin, (int)T, E_, H_);

  // 4. silu/gelu + coupled l2-normalize (in place)
  act_kernel<<<(int)T, 256, 0, stream>>>(qkv);

  // 5. depthwise conv + beta gate + f16 pack (q row-major, k/v transposed)
  conv_pack_kernel<<<dim3(L_, B_), 256, 0, stream>>>(qkv, betaLin, conv_w,
                                                     qF16, kT, vT);

  // 6. state[b] = v_new^T @ k  (E x E, K = L)
  for (int b = 0; b < B_; b++)
    gemm_wmma_kernel<<<dim3(E_ / 128, E_ / 128), 256, 0, stream>>>(
        vT + (size_t)b * E_ * L_, kT + (size_t)b * E_ * L_, nullptr, nullptr,
        stateF32 + (size_t)b * E_ * E_, E_, E_, L_);
  cast_f16_kernel<<<2048, 256, 0, stream>>>(stateF32, stateF16,
                                            (size_t)B_ * E_ * E_);

  // 7. attn[b] = q @ state^T  (L x E, K = E)
  for (int b = 0; b < B_; b++)
    gemm_wmma_kernel<<<dim3(L_ / 128, E_ / 128), 256, 0, stream>>>(
        qF16 + (size_t)b * L_ * E_, stateF16 + (size_t)b * E_ * E_, nullptr,
        nullptr, attn + (size_t)b * L_ * E_, L_, E_, E_);

  // 8. y = LN2(attn * attn_scale) (f16)
  ln_to_f16_kernel<<<(int)T, 256, 0, stream>>>(attn, ln2_g, ln2_b, a_scale,
                                               yF16, E_);

  // 9. x2 = x + y @ W_out^T + b_out
  gemm_wmma_kernel<<<dim3((int)T / 128, H_ / 128), 256, 0, stream>>>(
      yF16, WoH, b_out, x, x2, (int)T, H_, E_);

  // 10. hn = LN(x2) (f16)
  ln_to_f16_kernel<<<(int)T, 256, 0, stream>>>(x2, ln1_g, ln1_b, nullptr,
                                               hnF16, H_);

  // 11. m1 = hn @ W1^T + b1 ; gelu -> f16
  gemm_wmma_kernel<<<dim3((int)T / 128, (4 * H_) / 128), 256, 0, stream>>>(
      hnF16, W1H, b1, nullptr, m1, (int)T, 4 * H_, H_);
  gelu_f16_kernel<<<4096, 256, 0, stream>>>(m1, geluF16, T * (size_t)(4 * H_));

  // 12. out = x2 + gelu @ W2^T + b2
  gemm_wmma_kernel<<<dim3((int)T / 128, H_ / 128), 256, 0, stream>>>(
      geluF16, W2H, b2, x2, out, (int)T, H_, 4 * H_);
}
